// DecoderMulti_6786048328640
// MI455X (gfx1250) — compile-verified, matrix operand order checked by reading, not run on hardware
//
#include <hip/hip_runtime.h>
#include <hip/hip_bf16.h>

typedef __attribute__((ext_vector_type(16))) __bf16 v16bf;
typedef __attribute__((ext_vector_type(8)))  float  v8f;

constexpr int Bn = 256, Sn = 512, Hn = 512, En = 128, Vn = 30000, RIN = 641;

union Frag { v16bf v; unsigned int u[8]; };

// round-to-nearest-even f32 -> bf16, packed pair
__device__ __forceinline__ unsigned int pack2bf(float a, float b) {
  union { float f; unsigned int u; } ca, cb;
  ca.f = a; cb.f = b;
  unsigned int ua = (ca.u + 0x7FFFu + ((ca.u >> 16) & 1u)) >> 16;
  unsigned int ub = (cb.u + 0x7FFFu + ((cb.u >> 16) & 1u)) >> 16;
  return (ua & 0xFFFFu) | (ub << 16);
}

__device__ __forceinline__ v8f wmma_bf16(Frag a, Frag b, v8f c) {
  return __builtin_amdgcn_wmma_f32_16x16x32_bf16(
      false, a.v, false, b.v, (short)0, c, false, false);
}

// ---- fragment-layout packed storage: frag f = 32 lanes x 8 dwords = 256 dwords ----
__device__ __forceinline__ void store_frag(unsigned int* dst, int f, int lane, const Frag& fr) {
  unsigned int* p = dst + ((long long)f << 8) + (lane << 3);
  uint4 lo; lo.x = fr.u[0]; lo.y = fr.u[1]; lo.z = fr.u[2]; lo.w = fr.u[3];
  uint4 hi; hi.x = fr.u[4]; hi.y = fr.u[5]; hi.z = fr.u[6]; hi.w = fr.u[7];
  *(uint4*)p       = lo;
  *(uint4*)(p + 4) = hi;
}

__device__ __forceinline__ Frag load_frag(const unsigned int* base, int f, int lane) {
  const uint4* p = (const uint4*)(base + ((long long)f << 8) + (lane << 3));
  uint4 lo = p[0], hi = p[1];
  Frag fr;
  fr.u[0] = lo.x; fr.u[1] = lo.y; fr.u[2] = lo.z; fr.u[3] = lo.w;
  fr.u[4] = hi.x; fr.u[5] = hi.y; fr.u[6] = hi.z; fr.u[7] = hi.w;
  return fr;
}

// A fragment, 16x32 bf16 tile from row-major f32 source.
// lane l(0..15): M=l, K = k0 + {0..7,16..23}; lane 16+l: K = k0 + {8..15,24..31}
__device__ __forceinline__ Frag load_a_rm(const float* src, int lda, int row0, int k0, int lane) {
  Frag f;
  const float* p = src + (long long)(row0 + (lane & 15)) * lda + k0 + ((lane & 16) ? 8 : 0);
#pragma unroll
  for (int j = 0; j < 4; ++j) {
    f.u[j]     = pack2bf(p[2 * j],      p[2 * j + 1]);
    f.u[4 + j] = pack2bf(p[16 + 2 * j], p[17 + 2 * j]);
  }
  return f;
}

// B fragment, 32x16 bf16 tile; source row-major KxN f32: B[k][n] = src[k*ldn + n]
__device__ __forceinline__ Frag load_b_rm(const float* src, int ldn, int n0, int k0, int lane) {
  Frag f;
  const float* p = src + (long long)(k0 + ((lane & 16) ? 8 : 0)) * ldn + n0 + (lane & 15);
#pragma unroll
  for (int j = 0; j < 4; ++j) {
    f.u[j]     = pack2bf(p[(2 * j) * ldn],      p[(2 * j + 1) * ldn]);
    f.u[4 + j] = pack2bf(p[(16 + 2 * j) * ldn], p[(17 + 2 * j) * ldn]);
  }
  return f;
}

// B fragment from W^T: W row-major NxK: B[k][n] = W[n*ldw + k], zero beyond kmax
__device__ __forceinline__ Frag load_b_wt(const float* W, int ldw, int kmax, int n0, int k0, int lane) {
  Frag f;
  const float* p = W + (long long)(n0 + (lane & 15)) * ldw;
  int kb = k0 + ((lane & 16) ? 8 : 0);
#pragma unroll
  for (int j = 0; j < 4; ++j) {
    int ka = kb + 2 * j, kc = kb + 16 + 2 * j;
    float a0 = (ka     < kmax) ? p[ka]     : 0.f;
    float a1 = (ka + 1 < kmax) ? p[ka + 1] : 0.f;
    float c0 = (kc     < kmax) ? p[kc]     : 0.f;
    float c1 = (kc + 1 < kmax) ? p[kc + 1] : 0.f;
    f.u[j]     = pack2bf(a0, a1);
    f.u[4 + j] = pack2bf(c0, c1);
  }
  return f;
}

// ---------------- one-time packing kernels (f32 -> bf16 fragment layout) ----------------
__global__ void k_pack_b_rm(const float* __restrict__ src, int ldn, int nfrags, int ktiles,
                            unsigned int* __restrict__ dst) {
  int gid = blockIdx.x * 256 + threadIdx.x;
  int f = gid >> 5, lane = gid & 31;
  if (f >= nfrags) return;
  int nt = f / ktiles, kt = f - nt * ktiles;
  store_frag(dst, f, lane, load_b_rm(src, ldn, nt * 16, kt * 32, lane));
}

__global__ void k_pack_b_wt(const float* __restrict__ W, int ldw, int kmax, int nfrags, int ktiles,
                            unsigned int* __restrict__ dst) {
  int gid = blockIdx.x * 256 + threadIdx.x;
  int f = gid >> 5, lane = gid & 31;
  if (f >= nfrags) return;
  int nt = f / ktiles, kt = f - nt * ktiles;
  store_frag(dst, f, lane, load_b_wt(W, ldw, kmax, nt * 16, kt * 32, lane));
}

__global__ void k_pack_a_rm(const float* __restrict__ src, int lda, int nfrags, int ktiles,
                            unsigned int* __restrict__ dst) {
  int gid = blockIdx.x * 256 + threadIdx.x;
  int f = gid >> 5, lane = gid & 31;
  if (f >= nfrags) return;
  int mt = f / ktiles, kt = f - mt * ktiles;
  store_frag(dst, f, lane, load_a_rm(src, lda, mt * 16, kt * 32, lane));
}

// ---------------- hid_part = h0 @ attn_w[:H] + attn_b  (B x H) ----------------
__global__ void k_hidpart(const float* __restrict__ h0, const float* __restrict__ attn_w,
                          const float* __restrict__ attn_b, float* __restrict__ hid_part) {
  __shared__ float hrow[Hn];
  int b = blockIdx.x >> 1;
  int k = ((blockIdx.x & 1) << 8) + threadIdx.x;
  hrow[threadIdx.x]       = h0[b * Hn + threadIdx.x];
  hrow[threadIdx.x + 256] = h0[b * Hn + threadIdx.x + 256];
  __syncthreads();
  float acc = attn_b[k];
  for (int j = 0; j < Hn; ++j) acc += hrow[j] * attn_w[j * Hn + k];
  hid_part[b * Hn + k] = acc;
}

// ------- fused: energy GEMM (WMMA bf16) + tanh + v_w reduce + mask -> scores(B,S) -------
// A (encoder rows) converted once, register-resident across all 32 N-tiles;
// B is the pre-packed w2 (pure b128 loads, L2-resident).
// launch_bounds(256,1): allow full VGPR budget so the 16 A fragments don't spill.
__global__ void __launch_bounds__(256, 1)
k_scores(const float* __restrict__ enc, const unsigned int* __restrict__ w2p,
         const float* __restrict__ hid_part, const float* __restrict__ v_w,
         const int* __restrict__ attn_mask, float* __restrict__ scores) {
  int lane = threadIdx.x & 31;
  int tile = blockIdx.x * 8 + (threadIdx.x >> 5);
  if (tile >= (Bn * Sn) / 16) return;
  int m0 = tile * 16;

  Frag afr[16];
#pragma unroll
  for (int kt = 0; kt < 16; ++kt) afr[kt] = load_a_rm(enc, Hn, m0, kt * 32, lane);

  float part[8];
#pragma unroll
  for (int r = 0; r < 8; ++r) part[r] = 0.f;

  int rbase = (lane & 16) ? 8 : 0;
  int cl = lane & 15;

#pragma unroll 1
  for (int nt = 0; nt < Hn / 16; ++nt) {
    v8f acc = {};
#pragma unroll
    for (int kt = 0; kt < 16; ++kt)
      acc = wmma_bf16(afr[kt], load_frag(w2p, nt * 16 + kt, lane), acc);
    int col = nt * 16 + cl;
    float vw = v_w[col];
#pragma unroll
    for (int r = 0; r < 8; ++r) {
      int bb = (m0 + rbase + r) & (Bn - 1);
      part[r] += vw * tanhf(acc[r] + hid_part[bb * Hn + col]);
    }
  }
#pragma unroll
  for (int r = 0; r < 8; ++r) {
    part[r] += __shfl_xor(part[r], 1, 32);
    part[r] += __shfl_xor(part[r], 2, 32);
    part[r] += __shfl_xor(part[r], 4, 32);
    part[r] += __shfl_xor(part[r], 8, 32);
  }
  if (cl < 8) {
    int m = m0 + rbase + cl;
    int bb = m & (Bn - 1);
    int s = m >> 8;                       // B == 256
    float sc = part[cl];
    if (attn_mask[bb * Sn + s] == 0) sc = -1e10f;
    scores[bb * Sn + s] = sc;
  }
}

// ---------------- softmax over S, in place ----------------
__global__ void k_softmax_s(float* __restrict__ sc) {
  int b = blockIdx.x, t = threadIdx.x;
  __shared__ float red[256];
  float s0 = sc[b * Sn + t], s1 = sc[b * Sn + t + 256];
  red[t] = fmaxf(s0, s1); __syncthreads();
  for (int w = 128; w > 0; w >>= 1) { if (t < w) red[t] = fmaxf(red[t], red[t + w]); __syncthreads(); }
  float mx = red[0]; __syncthreads();
  float e0 = __expf(s0 - mx), e1 = __expf(s1 - mx);
  red[t] = e0 + e1; __syncthreads();
  for (int w = 128; w > 0; w >>= 1) { if (t < w) red[t] += red[t + w]; __syncthreads(); }
  float inv = 1.f / red[0];
  sc[b * Sn + t] = e0 * inv; sc[b * Sn + t + 256] = e1 * inv;
}

// ---------------- weighted[b,h] = sum_s a[b,s] * enc[s,b,h] ----------------
__global__ void k_weighted(const float* __restrict__ a, const float* __restrict__ enc,
                           float* __restrict__ weighted) {
  __shared__ float arow[Sn];
  int b = blockIdx.x >> 1;
  int h = ((blockIdx.x & 1) << 8) + threadIdx.x;
  arow[threadIdx.x]       = a[b * Sn + threadIdx.x];
  arow[threadIdx.x + 256] = a[b * Sn + threadIdx.x + 256];
  __syncthreads();
  float acc = 0.f;
  for (int s = 0; s < Sn; ++s)
    acc += arow[s] * enc[((long long)s * Bn + b) * Hn + h];
  weighted[b * Hn + h] = acc;
}

// ---------------- x = [weighted | embedded | rate | zero-pad] (B x 672) ----------------
__global__ void k_buildx(const float* __restrict__ weighted, const float* __restrict__ emb_id,
                         const int* __restrict__ input_id, const float* __restrict__ input_rate,
                         float* __restrict__ xpad) {
  int idx = blockIdx.x * 256 + threadIdx.x;
  int b = idx / 672, j = idx - b * 672;
  float v;
  if (j < Hn)            v = weighted[b * Hn + j];
  else if (j < Hn + En)  v = emb_id[(long long)input_id[b] * En + (j - Hn)];
  else if (j == Hn + En) v = input_rate[b];
  else                   v = 0.f;
  xpad[idx] = v;
}

// -------- GEMM, both operands pre-packed bf16 fragments: b128 loads + wmma only --------
__global__ void __launch_bounds__(256)
k_gemm_pp(const unsigned int* __restrict__ Ap,
          const unsigned int* __restrict__ Bp,
          const float* __restrict__ bias,
          float* __restrict__ C, int N, int ktiles, int relu) {
  int lane = threadIdx.x & 31;
  int tile = blockIdx.x * 8 + (threadIdx.x >> 5);
  int ntiles = N >> 4;
  if (tile >= (Bn >> 4) * ntiles) return;
  int mt = tile / ntiles, nt = tile - mt * ntiles;
  const unsigned int* ab = Ap + (long long)mt * ktiles * 256;
  const unsigned int* bb = Bp + (long long)nt * ktiles * 256;
  v8f acc = {};
  for (int kt = 0; kt < ktiles; ++kt)
    acc = wmma_bf16(load_frag(ab, kt, lane), load_frag(bb, kt, lane), acc);
  int col = nt * 16 + (lane & 15);
  int rbase = (lane & 16) ? 8 : 0;
  int m0 = mt * 16;
  float bv = bias[col];
#pragma unroll
  for (int r = 0; r < 8; ++r) {
    float o = acc[r] + bv;
    if (relu) o = fmaxf(o, 0.f);
    C[(long long)(m0 + rbase + r) * N + col] = o;
  }
}

// ---------------- GRU gate math -> h1 ----------------
__global__ void k_gru(const float* __restrict__ gi, const float* __restrict__ gh,
                      const float* __restrict__ h0, float* __restrict__ h1,
                      float* __restrict__ out_h1) {
  int idx = blockIdx.x * 256 + threadIdx.x;
  int b = idx >> 9, h = idx & (Hn - 1);
  const float* gib = gi + b * 3 * Hn;
  const float* ghb = gh + b * 3 * Hn;
  float r = 1.f / (1.f + __expf(-(gib[h] + ghb[h])));
  float z = 1.f / (1.f + __expf(-(gib[Hn + h] + ghb[Hn + h])));
  float n = tanhf(gib[2 * Hn + h] + r * ghb[2 * Hn + h]);
  float hv = (1.f - z) * n + z * h0[idx];
  h1[idx] = hv;
  out_h1[idx] = hv;
}

// ---- constrained softmax over V + clip + log, IN PLACE on logits(=out_id); argmax id ----
__global__ void k_csoftmax(float* __restrict__ logits, const float* __restrict__ cons,
                           int* __restrict__ maxid) {
  int b = blockIdx.x, t = threadIdx.x;
  __shared__ float sv[256];
  __shared__ int   si[256];
  float* lr = logits + (long long)b * Vn;
  const float* cr = cons + (long long)b * Vn;
  float mx = -3.4e38f;
  for (int v = t; v < Vn; v += 256) mx = fmaxf(mx, lr[v]);
  sv[t] = mx; __syncthreads();
  for (int w = 128; w > 0; w >>= 1) { if (t < w) sv[t] = fmaxf(sv[t], sv[t + w]); __syncthreads(); }
  mx = sv[0]; __syncthreads();

  float sum = 0.f, bv = -1.f; int bi = Vn;
  for (int v = t; v < Vn; v += 256) {
    float e = __expf(lr[v] - mx) * cr[v];
    sum += e;
    if (e > bv) { bv = e; bi = v; }
  }
  sv[t] = sum; __syncthreads();
  for (int w = 128; w > 0; w >>= 1) { if (t < w) sv[t] += sv[t + w]; __syncthreads(); }
  float inv = 1.f / sv[0]; __syncthreads();

  sv[t] = bv; si[t] = bi; __syncthreads();
  for (int w = 128; w > 0; w >>= 1) {
    if (t < w) {
      if (sv[t + w] > sv[t] || (sv[t + w] == sv[t] && si[t + w] < si[t])) {
        sv[t] = sv[t + w]; si[t] = si[t + w];
      }
    }
    __syncthreads();
  }
  if (t == 0) maxid[b] = si[0];

  for (int v = t; v < Vn; v += 256) {   // read-before-write per element, same thread
    float e = __expf(lr[v] - mx) * cr[v];
    float p = fminf(fmaxf(e * inv, 1e-6f), 1.0f);
    lr[v] = __logf(p);
  }
}

// ---------------- cat = [id_emb | h1]  (B x 640) ----------------
__global__ void k_buildcat(const float* __restrict__ emb_id, const int* __restrict__ maxid,
                           const float* __restrict__ h1, float* __restrict__ cat) {
  int idx = blockIdx.x * 256 + threadIdx.x;
  int b = idx / 640, j = idx - b * 640;
  cat[idx] = (j < En) ? emb_id[(long long)maxid[b] * En + j] : h1[b * Hn + (j - En)];
}

// ---------------- prediction_rate = sigmoid(rate_in . rate_w + rate_b) ----------------
__global__ void k_rate(const float* __restrict__ rate_in, const float* __restrict__ rate_w,
                       const float* __restrict__ rate_b, float* __restrict__ out_rate) {
  int b = blockIdx.x, t = threadIdx.x;
  __shared__ float red[256];
  float v = rate_in[b * Hn + t] * rate_w[t] + rate_in[b * Hn + t + 256] * rate_w[t + 256];
  red[t] = v; __syncthreads();
  for (int w = 128; w > 0; w >>= 1) { if (t < w) red[t] += red[t + w]; __syncthreads(); }
  if (t == 0) out_rate[b] = 1.f / (1.f + __expf(-(red[0] + rate_b[0])));
}

extern "C" void kernel_launch(void* const* d_in, const int* in_sizes, int n_in,
                              void* d_out, int out_size, void* d_ws, size_t ws_size,
                              hipStream_t stream) {
  const int*   input_id   = (const int*)  d_in[0];
  const float* input_rate = (const float*)d_in[1];
  const float* hidden     = (const float*)d_in[2];   // (1,B,H)
  const float* enc        = (const float*)d_in[3];   // (S,B,H)
  const int*   attn_mask  = (const int*)  d_in[4];
  const float* cons       = (const float*)d_in[5];
  const float* emb_id     = (const float*)d_in[6];
  const float* attn_w     = (const float*)d_in[7];
  const float* attn_b     = (const float*)d_in[8];
  const float* v_w        = (const float*)d_in[9];
  const float* w_ih       = (const float*)d_in[10];
  const float* w_hh       = (const float*)d_in[11];
  const float* b_ih       = (const float*)d_in[12];
  const float* b_hh       = (const float*)d_in[13];
  const float* fc_id_w    = (const float*)d_in[14];
  const float* fc_id_b    = (const float*)d_in[15];
  const float* tandem_w   = (const float*)d_in[16];
  const float* tandem_b   = (const float*)d_in[17];
  const float* rate_w     = (const float*)d_in[18];
  const float* rate_b     = (const float*)d_in[19];

  float* ws       = (float*)d_ws;
  float* hid_part = ws;                         // 131072
  float* scores   = ws + 131072;                // 131072 (becomes attn probs in place)
  float* weighted = ws + 262144;                // 131072
  float* xpad     = ws + 393216;                // B*672 = 172032
  float* gi       = ws + 565248;                // 393216
  float* gh       = ws + 958464;                // 393216
  float* h1       = ws + 1351680;               // 131072
  float* catb     = ws + 1482752;               // B*640 = 163840
  float* rate_in  = ws + 1646592;               // 131072
  int*   maxid    = (int*)(ws + 1777664);       // 256
  unsigned int* w2p   = (unsigned int*)(ws + 1777920);   // 32*16*256   = 131072
  unsigned int* wihp  = (unsigned int*)(ws + 1908992);   // 96*21*256   = 516096
  unsigned int* whhp  = (unsigned int*)(ws + 2425088);   // 96*16*256   = 393216
  unsigned int* fcp   = (unsigned int*)(ws + 2818304);   // 1875*16*256 = 7680000
  unsigned int* tdp   = (unsigned int*)(ws + 10498304);  // 32*20*256   = 163840
  unsigned int* xpadp = (unsigned int*)(ws + 10662144);  // 16*21*256   = 86016
  unsigned int* h1p   = (unsigned int*)(ws + 10748160);  // 16*16*256   = 65536
  unsigned int* catp  = (unsigned int*)(ws + 10813696);  // 16*20*256   = 81920
  unsigned int* h0p   = (unsigned int*)(ws + 10895616);  // 16*16*256   = 65536
  // total: 10961152 floats (~43.9 MB)

  float* out_id   = (float*)d_out;                       // B*V (doubles as logits buffer)
  float* out_rate = out_id + (long long)Bn * Vn;         // B
  float* out_h1   = out_rate + Bn;                       // B*H

  // ---- one-time weight packing (f32 -> bf16 WMMA fragments) ----
  k_pack_b_rm<<<64,   256, 0, stream>>>(attn_w + Hn * Hn, Hn, 32 * 16,   16, w2p);
  k_pack_b_wt<<<252,  256, 0, stream>>>(w_ih, RIN, RIN,   96 * 21,       21, wihp);
  k_pack_b_wt<<<192,  256, 0, stream>>>(w_hh, Hn, Hn,     96 * 16,       16, whhp);
  k_pack_b_rm<<<3750, 256, 0, stream>>>(fc_id_w, Vn,      1875 * 16,     16, fcp);
  k_pack_b_rm<<<80,   256, 0, stream>>>(tandem_w, Hn,     32 * 20,       20, tdp);
  k_pack_a_rm<<<32,   256, 0, stream>>>(hidden, Hn,       16 * 16,       16, h0p);

  // ---- attention ----
  k_hidpart  <<<512,  256, 0, stream>>>(hidden, attn_w, attn_b, hid_part);
  k_scores   <<<1024, 256, 0, stream>>>(enc, w2p, hid_part, v_w, attn_mask, scores);
  k_softmax_s<<<256,  256, 0, stream>>>(scores);
  k_weighted <<<512,  256, 0, stream>>>(scores, enc, weighted);

  // ---- GRU ----
  k_buildx   <<<672,  256, 0, stream>>>(weighted, emb_id, input_id, input_rate, xpad);
  k_pack_a_rm<<<42,   256, 0, stream>>>(xpad, 672, 16 * 21, 21, xpadp);
  k_gemm_pp  <<<192,  256, 0, stream>>>(xpadp, wihp, b_ih, gi, 3 * Hn, 21, 0);
  k_gemm_pp  <<<192,  256, 0, stream>>>(h0p,   whhp, b_hh, gh, 3 * Hn, 16, 0);
  k_gru      <<<512,  256, 0, stream>>>(gi, gh, hidden, h1, out_h1);

  // ---- vocab head (logits straight into d_out, softmax in place) ----
  k_pack_a_rm<<<32,   256, 0, stream>>>(h1, Hn, 16 * 16, 16, h1p);
  k_gemm_pp  <<<3750, 256, 0, stream>>>(h1p, fcp, fc_id_b, out_id, Vn, 16, 0);
  k_csoftmax <<<256,  256, 0, stream>>>(out_id, cons, maxid);

  // ---- rate head ----
  k_buildcat <<<640,  256, 0, stream>>>(emb_id, maxid, h1, catb);
  k_pack_a_rm<<<40,   256, 0, stream>>>(catb, 640, 16 * 20, 20, catp);
  k_gemm_pp  <<<64,   256, 0, stream>>>(catp, tdp, tandem_b, rate_in, Hn, 20, 1);
  k_rate     <<<256,  256, 0, stream>>>(rate_in, rate_w, rate_b, out_rate);
}
